// HumanVAttention_45200235823699
// MI455X (gfx1250) — matrix operand order repaired and checked by reading, not compiled, not run on hardware
//
#include <hip/hip_runtime.h>

// ---------------- problem constants ----------------
#define Tq    2048
#define HID_  2048
#define NH    32
#define NKVH  8
#define DH    64
#define GQ    4
#define KVD   (NKVH * DH)     // 512
#define HD_   (NH * DH)       // 2048
#define SCALE 0.125f          // 1/sqrt(64)

typedef _Float16 h8  __attribute__((ext_vector_type(8)));
typedef _Float16 h16 __attribute__((ext_vector_type(16)));
typedef _Float16 h4  __attribute__((ext_vector_type(4)));
typedef float    f8v __attribute__((ext_vector_type(8)));

#if defined(__gfx1250__) && __has_builtin(__builtin_amdgcn_tensor_load_to_lds)
#define HAVE_TDM 1
typedef unsigned int u32x4 __attribute__((ext_vector_type(4)));
typedef int          i32x4v __attribute__((ext_vector_type(4)));
typedef int          i32x8v __attribute__((ext_vector_type(8)));
#else
#define HAVE_TDM 0
#endif

static __device__ __forceinline__ h16 pack16(h8 lo, h8 hi) {
    return __builtin_shufflevector(lo, hi, 0,1,2,3,4,5,6,7,8,9,10,11,12,13,14,15);
}

static __device__ __forceinline__ f8v wmma16(h16 a, h16 b, f8v c) {
    // D = A(16x32 f16) * B(32x16 f16) + C(16x16 f32)
    return __builtin_amdgcn_wmma_f32_16x16x32_f16(false, a, false, b, (short)0, c, false, false);
}

// ---------------- f32 -> f16 convert ----------------
__global__ void f32_to_f16_kernel(const float* __restrict__ s, _Float16* __restrict__ d, int n) {
    int i = (blockIdx.x * blockDim.x + threadIdx.x) * 4;
    if (i + 3 < n) {
        float4 v = *(const float4*)(s + i);
        h4 h = { (_Float16)v.x, (_Float16)v.y, (_Float16)v.z, (_Float16)v.w };
        *(h4*)(d + i) = h;
    }
}

// ---------------- tiled WMMA GEMM:  C[M,N] = A[M,K] @ W[N,K]^T ----------------
#define BM 128
#define BN 128
#define BK 64
#define LDT 72   // BK + 8 halves pad (144 B rows; TDM pad reproduces this)

#if HAVE_TDM
// Issue one TDM 2D tile load: BM rows x BK f16 elems, row stride K elems,
// LDS rows padded 128B + 16B -> 144B (matches LDT).
static __device__ __forceinline__ void tdm_load_tile(const _Float16* gsrc,
                                                     unsigned int lds_off, int K) {
    unsigned long long ga = (unsigned long long)(uintptr_t)gsrc;
    u32x4 g0;
    g0[0] = 1u;                                   // count=1 (valid user descriptor)
    g0[1] = lds_off;                              // lds_addr (bytes)
    g0[2] = (unsigned int)ga;                     // global_addr[31:0]
    g0[3] = (unsigned int)(ga >> 32) | (2u << 30);// global_addr[56:32] | type=2
    i32x8v g1;
    g1[0] = (1 << 16)      // data_size = 2 bytes
          | (1 << 20)      // pad_enable
          | (4 << 22)      // pad_interval: 32 DWORDs (=128B = one tile row)
          | (3 << 25);     // pad_amount: 4 DWORDs (=16B) -> 144B row stride
    g1[1] = (K & 0xFFFF) << 16;                         // tensor_dim0 lo16
    g1[2] = ((K >> 16) & 0xFFFF);                       // tensor_dim0 hi16 | dim1 lo16(0)
    g1[3] = 16 | (BK << 16);                            // tensor_dim1 hi16 (1<<20) | tile_dim0
    g1[4] = BM;                                         // tile_dim1 | tile_dim2=0
    g1[5] = K;                                          // tensor_dim0_stride lo32
    g1[6] = 0;                                          // stride hi | dim1_stride lo
    g1[7] = 0;
    i32x4v z4 = {0, 0, 0, 0};
    i32x8v z8 = {0, 0, 0, 0, 0, 0, 0, 0};
    __builtin_amdgcn_tensor_load_to_lds(g0, g1, z4, z4, z8, 0);
}
static __device__ __forceinline__ unsigned int lds_byte_off(const void* p) {
    // low 32 bits of a flat pointer into the LDS aperture are the LDS byte offset
    return (unsigned int)(uintptr_t)p;
}
#endif

template <bool OUT16, bool TRANSB>
__global__ __launch_bounds__(256) void gemm_wmma_kernel(
    const _Float16* __restrict__ A, const _Float16* __restrict__ W,
    _Float16* __restrict__ C16, float* __restrict__ C32,
    int M, int N, int K)
{
    __shared__ __align__(16) _Float16 As[2][BM * LDT];
    __shared__ __align__(16) _Float16 Bs[2][BM * LDT];

    const int tid  = threadIdx.x;
    const int wave = tid >> 5;
    const int lane = tid & 31;
    const int wrow = wave & 3;   // 4 wave-rows * 32 rows
    const int wcol = wave >> 2;  // 2 wave-cols * 64 cols
    const int row0 = blockIdx.y * BM;
    const int col0 = blockIdx.x * BN;

    const int hi16 = lane >> 4;
    const int l15  = lane & 15;
    const int akb  = hi16 << 3;  // A-operand K base within 32-chunk: 0 or 8
    const int bkb  = hi16 << 4;  // unused; B loads use akb (same k order as A)

    f8v zero = {0.f,0.f,0.f,0.f,0.f,0.f,0.f,0.f};
    f8v acc[2][4];
    #pragma unroll
    for (int i = 0; i < 2; ++i)
        #pragma unroll
        for (int j = 0; j < 4; ++j) acc[i][j] = zero;

    const int nsteps = K / BK;

#if HAVE_TDM
    if (tid < 32) {  // wave 0 drives the Tensor Data Mover
        tdm_load_tile(A + (size_t)row0 * K, lds_byte_off(&As[0][0]), K);
        tdm_load_tile(W + (size_t)col0 * K, lds_byte_off(&Bs[0][0]), K);
    }
#else
    const int ldr = tid >> 3;           // 0..31
    const int ldc = (tid & 7) << 3;     // halves within BK
#endif

    for (int s = 0; s < nsteps; ++s) {
        const int cur = s & 1;
#if HAVE_TDM
        if (tid < 32) {
            if (s + 1 < nsteps) {
                tdm_load_tile(A + (size_t)row0 * K + (s + 1) * BK,
                              lds_byte_off(&As[cur ^ 1][0]), K);
                tdm_load_tile(W + (size_t)col0 * K + (s + 1) * BK,
                              lds_byte_off(&Bs[cur ^ 1][0]), K);
                __builtin_amdgcn_s_wait_tensorcnt(2);  // current tile complete
            } else {
                __builtin_amdgcn_s_wait_tensorcnt(0);
            }
        }
        __syncthreads();
#else
        #pragma unroll
        for (int i = 0; i < 4; ++i) {
            int r = ldr + i * 32;
            *(uint4*)(&As[cur][r * LDT + ldc]) =
                *(const uint4*)(A + (size_t)(row0 + r) * K + s * BK + ldc);
            *(uint4*)(&Bs[cur][r * LDT + ldc]) =
                *(const uint4*)(W + (size_t)(col0 + r) * K + s * BK + ldc);
        }
        __syncthreads();
#endif
        // A fragments: 2 row-tiles x 2 k-chunks
        h16 af[2][2];
        #pragma unroll
        for (int i = 0; i < 2; ++i)
            #pragma unroll
            for (int kk = 0; kk < 2; ++kk) {
                const _Float16* p = &As[cur][(wrow * 32 + i * 16 + l15) * LDT + kk * 32 + akb];
                af[i][kk] = pack16(*(const h8*)p, *(const h8*)(p + 16));
            }
        #pragma unroll
        for (int j = 0; j < 4; ++j) {
            h16 bf[2];
            #pragma unroll
            for (int kk = 0; kk < 2; ++kk) {
                const _Float16* p = &Bs[cur][(wcol * 64 + j * 16 + l15) * LDT + kk * 32 + akb];
                bf[kk] = pack16(*(const h8*)p, *(const h8*)(p + 16));
            }
            #pragma unroll
            for (int i = 0; i < 2; ++i)
                #pragma unroll
                for (int kk = 0; kk < 2; ++kk)
                    acc[i][j] = wmma16(af[i][kk], bf[kk], acc[i][j]);
        }
        __syncthreads();
    }

    // store C (C-layout: lane = col, vgpr r = row r + 8*hi16)
    #pragma unroll
    for (int i = 0; i < 2; ++i)
        #pragma unroll
        for (int j = 0; j < 4; ++j)
            #pragma unroll
            for (int r = 0; r < 8; ++r) {
                int gr = row0 + wrow * 32 + i * 16 + (hi16 << 3) + r;
                int gc = col0 + wcol * 64 + j * 16 + l15;
                if (OUT16) {
                    if (TRANSB) C16[(size_t)gc * M + gr] = (_Float16)acc[i][j][r];
                    else        C16[(size_t)gr * N + gc] = (_Float16)acc[i][j][r];
                } else {
                    C32[(size_t)gr * N + gc] = acc[i][j][r];
                }
            }
}

// ---------------- RoPE (in place, fp32 math on f16 buffer) ----------------
__global__ void rope_kernel(_Float16* __restrict__ P,
                            const float* __restrict__ cosb,
                            const float* __restrict__ sinb, int nh)
{
    int idx = blockIdx.x * blockDim.x + threadIdx.x;
    int d = idx & 31;
    int h = (idx >> 5) % nh;
    int t = idx / (nh << 5);
    if (t >= Tq) return;
    _Float16* p = P + ((size_t)t * nh + h) * DH;
    float x1 = (float)p[d];
    float x2 = (float)p[d + 32];
    float c1 = cosb[t * DH + d],      s1 = sinb[t * DH + d];
    float c2 = cosb[t * DH + d + 32], s2 = sinb[t * DH + d + 32];
    p[d]      = (_Float16)(x1 * c1 - x2 * s1);
    p[d + 32] = (_Float16)(x2 * c2 + x1 * s2);
}

// ---------------- flash attention, causal, GQA ----------------
template <bool MASKED>
static __device__ __forceinline__ void attn_tile(
    int kb0, int q0, int hi16, int l15, int akb, int bkb, int kvh,
    const h16 (&qf)[2],
    const _Float16* __restrict__ Km, const _Float16* __restrict__ Vt,
    float (&mrow)[8], float (&lrow)[8], f8v (&oacc)[4],
    _Float16* Pw)
{
    f8v zero = {0.f,0.f,0.f,0.f,0.f,0.f,0.f,0.f};
    f8v s0 = zero, s1 = zero;
    #pragma unroll
    for (int dk = 0; dk < 2; ++dk) {
        const _Float16* kp0 = Km + (size_t)(kb0 + l15) * KVD + kvh * DH + dk * 32 + akb;
        s0 = wmma16(qf[dk], pack16(*(const h8*)kp0, *(const h8*)(kp0 + 16)), s0);
        const _Float16* kp1 = Km + (size_t)(kb0 + 16 + l15) * KVD + kvh * DH + dk * 32 + akb;
        s1 = wmma16(qf[dk], pack16(*(const h8*)kp1, *(const h8*)(kp1 + 16)), s1);
    }
    // online softmax; a row lives across 16 lanes (C-layout)
    #pragma unroll
    for (int r = 0; r < 8; ++r) {
        float v0 = s0[r] * SCALE;
        float v1 = s1[r] * SCALE;
        bool ok0 = true, ok1 = true;
        if (MASKED) {
            const int qi = q0 + (hi16 << 3) + r;
            ok0 = (kb0 + l15)      <= qi;
            ok1 = (kb0 + 16 + l15) <= qi;
        }
        float mx = MASKED ? fmaxf(ok0 ? v0 : -3.0e38f, ok1 ? v1 : -3.0e38f)
                          : fmaxf(v0, v1);
        #pragma unroll
        for (int off = 8; off >= 1; off >>= 1) mx = fmaxf(mx, __shfl_xor(mx, off, 32));
        float mnew  = fmaxf(mrow[r], mx);
        float alpha = __expf(mrow[r] - mnew);
        float e0 = MASKED ? (ok0 ? __expf(v0 - mnew) : 0.f) : __expf(v0 - mnew);
        float e1 = MASKED ? (ok1 ? __expf(v1 - mnew) : 0.f) : __expf(v1 - mnew);
        float sm = e0 + e1;
        #pragma unroll
        for (int off = 8; off >= 1; off >>= 1) sm += __shfl_xor(sm, off, 32);
        lrow[r] = lrow[r] * alpha + sm;
        mrow[r] = mnew;
        #pragma unroll
        for (int j = 0; j < 4; ++j) oacc[j][r] *= alpha;
        Pw[((hi16 << 3) + r) * 40 + l15]      = (_Float16)e0;
        Pw[((hi16 << 3) + r) * 40 + 16 + l15] = (_Float16)e1;
    }
    // reload P as A-operand fragment (16x32)
    const _Float16* pr = Pw + l15 * 40 + akb;
    h16 pf = pack16(*(const h8*)pr, *(const h8*)(pr + 16));
    // O += P @ V : B-operand straight from transposed V (contiguous per lane)
    #pragma unroll
    for (int j = 0; j < 4; ++j) {
        const _Float16* vp = Vt + (size_t)(kvh * DH + j * 16 + l15) * Tq + kb0 + akb;
        oacc[j] = wmma16(pf, pack16(*(const h8*)vp, *(const h8*)(vp + 16)), oacc[j]);
    }
}

// grid (Tq/64, NH), block 128 (4 waves); wave owns 16 query rows of one head
__global__ __launch_bounds__(128) void attn_wmma_kernel(
    const _Float16* __restrict__ Q,    // [Tq, HD_]
    const _Float16* __restrict__ Km,   // [Tq, KVD]
    const _Float16* __restrict__ Vt,   // [KVD, Tq]  (transposed V)
    _Float16* __restrict__ AO)         // [Tq, HD_]
{
    __shared__ __align__(16) _Float16 Pl[4][16 * 40];
    const int h    = blockIdx.y;
    const int kvh  = h >> 2;           // h / GQ
    const int wave = threadIdx.x >> 5;
    const int lane = threadIdx.x & 31;
    const int q0   = blockIdx.x * 64 + wave * 16;
    const int hi16 = lane >> 4;
    const int l15  = lane & 15;
    const int akb  = hi16 << 3;
    const int bkb  = hi16 << 4;

    // Q tile 16x64 as two A-operand fragments (k-chunks of 32)
    h16 qf[2];
    {
        const _Float16* qrow = Q + (size_t)(q0 + l15) * HD_ + h * DH;
        #pragma unroll
        for (int dk = 0; dk < 2; ++dk) {
            const _Float16* p = qrow + dk * 32 + akb;
            qf[dk] = pack16(*(const h8*)p, *(const h8*)(p + 16));
        }
    }

    float mrow[8], lrow[8];
    #pragma unroll
    for (int r = 0; r < 8; ++r) { mrow[r] = -3.0e38f; lrow[r] = 0.f; }
    f8v zero = {0.f,0.f,0.f,0.f,0.f,0.f,0.f,0.f};
    f8v oacc[4];
    #pragma unroll
    for (int j = 0; j < 4; ++j) oacc[j] = zero;

    _Float16* Pw = &Pl[wave][0];

    // fully-unmasked tiles: all keys <= every query row of this wave
    const int kfull = q0 >> 5;
    for (int kt = 0; kt < kfull; ++kt)
        attn_tile<false>(kt * 32, q0, hi16, l15, akb, bkb, kvh, qf, Km, Vt,
                         mrow, lrow, oacc, Pw);
    // exactly one diagonal tile needs causal masking (16 rows fit in one 32-key tile)
    attn_tile<true>(kfull * 32, q0, hi16, l15, akb, bkb, kvh, qf, Km, Vt,
                    mrow, lrow, oacc, Pw);

    // normalize and store f16 attention output
    #pragma unroll
    for (int j = 0; j < 4; ++j)
        #pragma unroll
        for (int r = 0; r < 8; ++r) {
            int t = q0 + (hi16 << 3) + r;
            AO[(size_t)t * HD_ + h * DH + j * 16 + l15] = (_Float16)(oacc[j][r] / lrow[r]);
        }
}

// ---------------- host-side launch ----------------
extern "C" void kernel_launch(void* const* d_in, const int* in_sizes, int n_in,
                              void* d_out, int out_size, void* d_ws, size_t ws_size,
                              hipStream_t stream) {
    (void)in_sizes; (void)n_in; (void)out_size; (void)ws_size;
    const float* x    = (const float*)d_in[0];
    const float* cosb = (const float*)d_in[1];
    const float* sinb = (const float*)d_in[2];
    // d_in[3] = attention_mask_4d (causal) -- computed inline instead
    const float* Wq = (const float*)d_in[4];
    const float* Wk = (const float*)d_in[5];
    const float* Wv = (const float*)d_in[6];
    const float* Wo = (const float*)d_in[7];
    float* out = (float*)d_out;

    _Float16* ws = (_Float16*)d_ws;
    size_t o = 0;
    _Float16* x16  = ws + o; o += (size_t)Tq * HID_;     // 4M halves
    _Float16* Wq16 = ws + o; o += (size_t)HD_ * HID_;    // 4M
    _Float16* Wk16 = ws + o; o += (size_t)KVD * HID_;    // 1M
    _Float16* Wv16 = ws + o; o += (size_t)KVD * HID_;    // 1M
    _Float16* Wo16 = ws + o; o += (size_t)HID_ * HD_;    // 4M
    _Float16* Q16  = ws + o; o += (size_t)Tq * HD_;      // 4M
    _Float16* K16  = ws + o; o += (size_t)Tq * KVD;      // 1M
    _Float16* Vt16 = ws + o; o += (size_t)KVD * Tq;      // 1M (transposed)
    _Float16* AO16 = x16;   // x16 is dead after the QKV GEMMs -> reuse

    const int CT = 256;
    f32_to_f16_kernel<<<(Tq*HID_/4 + CT-1)/CT, CT, 0, stream>>>(x,  x16,  Tq*HID_);
    f32_to_f16_kernel<<<(HD_*HID_/4 + CT-1)/CT, CT, 0, stream>>>(Wq, Wq16, HD_*HID_);
    f32_to_f16_kernel<<<(KVD*HID_/4 + CT-1)/CT, CT, 0, stream>>>(Wk, Wk16, KVD*HID_);
    f32_to_f16_kernel<<<(KVD*HID_/4 + CT-1)/CT, CT, 0, stream>>>(Wv, Wv16, KVD*HID_);
    f32_to_f16_kernel<<<(HID_*HD_/4 + CT-1)/CT, CT, 0, stream>>>(Wo, Wo16, HID_*HD_);

    // Q = x @ Wq^T ; K = x @ Wk^T ; V^T stored directly
    gemm_wmma_kernel<true,  false><<<dim3(HD_/BN, Tq/BM), 256, 0, stream>>>(x16, Wq16, Q16,  nullptr, Tq, HD_, HID_);
    gemm_wmma_kernel<true,  false><<<dim3(KVD/BN, Tq/BM), 256, 0, stream>>>(x16, Wk16, K16,  nullptr, Tq, KVD, HID_);
    gemm_wmma_kernel<true,  true ><<<dim3(KVD/BN, Tq/BM), 256, 0, stream>>>(x16, Wv16, Vt16, nullptr, Tq, KVD, HID_);

    rope_kernel<<<(Tq*NH*32)/CT,   CT, 0, stream>>>(Q16, cosb, sinb, NH);
    rope_kernel<<<(Tq*NKVH*32)/CT, CT, 0, stream>>>(K16, cosb, sinb, NKVH);

    attn_wmma_kernel<<<dim3(Tq/64, NH), 128, 0, stream>>>(Q16, K16, Vt16, AO16);

    // out = AO @ Wo^T  (fp32 output)
    gemm_wmma_kernel<false, false><<<dim3(HID_/BN, Tq/BM), 256, 0, stream>>>(AO16, Wo16, nullptr, out, Tq, HID_, HID_);
}
